// CRF_36825049596026
// MI455X (gfx1250) — matrix operand order repaired and checked by reading, not compiled, not run on hardware
//
#include <hip/hip_runtime.h>
#include <hip/hip_bf16.h>

typedef __attribute__((ext_vector_type(16))) __bf16 v16bf;
typedef __attribute__((ext_vector_type(8)))  float  v8f;

union ABfrag { v16bf v; unsigned int u[8]; };

__device__ __forceinline__ unsigned short f2bf(float f) {
    union { float f; unsigned int u; } x; x.f = f;
    unsigned int u = x.u;
    // round-to-nearest-even fp32 -> bf16
    return (unsigned short)((u + 0x7FFFu + ((u >> 16) & 1u)) >> 16);
}

// ---------------------------------------------------------------------------
// Kernel 1: logits[M=32768, N=128] = x[M, K=1024] @ w[K, N] + bias
//           (+ left boundary at t==0 rows, + right boundary at t==511 rows)
// Block: 256 threads = 8 waves; each block owns a 16-row M strip, wave w
// computes the 16x16 tile at N = 16*w.  K advances in 64-wide super-chunks
// through ping-pong LDS buffers (A bf16 rows, B bf16 transposed): 2 wmma per
// iteration, one barrier per iteration, staging overlapped behind the wmma.
// ---------------------------------------------------------------------------
#define LDA 66   // ushort row stride for A  (16 rows x (64+2 pad))
#define LDB 66   // ushort row stride for Bt (128 rows x (64+2 pad))

__global__ __launch_bounds__(256) void crf_gemm_logits(
    const float* __restrict__ x, const float* __restrict__ w,
    const float* __restrict__ bias, const float* __restrict__ lb,
    const float* __restrict__ rb, float* __restrict__ logits)
{
    __shared__ unsigned short lsA[2][16 * LDA];     // A chunk: [m][k] bf16
    __shared__ unsigned short lsB[2][128 * LDB];    // B chunk transposed: [n][k]

    const int tid     = threadIdx.x;
    const int lane    = tid & 31;
    const int wave    = tid >> 5;          // tileN index 0..7
    const int rowBase = blockIdx.x * 16;

    const int m  = lane & 15;
    const int kh = lane >> 4;              // lane-half selects K sub-range
    const int n  = (wave << 4) + (lane & 15);

    v8f acc = {};

    auto stage = [&](int kc, int buf) {
        unsigned short* sa = &lsA[buf][0];
        unsigned short* sb = &lsB[buf][0];
        // A: 16 rows x 64 cols, coalesced 256B row reads; prefetch next chunk
        for (int i = tid; i < 1024; i += 256) {
            int r = i >> 6, c = i & 63;
            const float* gx = &x[(rowBase + r) * 1024 + kc + c];
            if (kc + 64 < 1024) __builtin_prefetch(gx + 64, 0, 0);
            sa[r * LDA + c] = f2bf(*gx);
        }
        // B: 64 k-rows x 128 n-cols, coalesced reads, transposed LDS store
        for (int i = tid; i < 8192; i += 256) {
            int r = i >> 7, c = i & 127;                // r = k, c = n
            sb[c * LDB + r] = f2bf(w[(kc + r) * 128 + c]);
        }
    };

    stage(0, 0);
    __syncthreads();

    for (int kc = 0; kc < 1024; kc += 64) {
        const int cur = (kc >> 6) & 1;
        const unsigned short* pa = &lsA[cur][m * LDA];
        const unsigned short* pb = &lsB[cur][n * LDB + (kh << 4)];

#pragma unroll
        for (int s = 0; s < 2; ++s) {                  // two 32-deep wmma steps
            ABfrag a, b;
#pragma unroll
            for (int v = 0; v < 8; ++v) {
                // A frag: k = s*32 + (lane>=16 ? 16 : 0) + 2v
                int kp = (s << 5) + (kh << 4) + (v << 1);
                a.u[v] = *(const unsigned int*)&pa[kp];
                // B frag: lane = col n; k = s*32 + kh*16 + 2v (+1 in hi half)
                b.u[v] = *(const unsigned int*)&pb[(s << 5) + (v << 1)];
            }
            acc = __builtin_amdgcn_wmma_f32_16x16x32_bf16(
                      false, a.v, false, b.v, (short)0, acc, false, false);
        }

        if (kc + 64 < 1024) stage(kc + 64, cur ^ 1);   // overlap behind wmma
        __syncthreads();
    }

    // Epilogue: C layout — VGPR v: lanes 0-15 -> M=v, lanes 16-31 -> M=8+v
    const float bn = bias[n], lbn = lb[n], rbn = rb[n];
#pragma unroll
    for (int v = 0; v < 8; ++v) {
        int mm = ((lane >> 4) << 3) + v;
        int gr = rowBase + mm;                 // global row = b*512 + t
        float val = acc[v] + bn;
        int t = gr & 511;
        if (t == 0)   val += lbn;
        if (t == 511) val += rbn;
        logits[gr * 128 + n] = val;
    }
}

// ---------------------------------------------------------------------------
// Kernel 2: Viterbi forward.  One block per batch, 512 threads = 4 sub-groups
// of 128; sub-group q reduces i in [32q, 32q+32).  trans (64 KB) lives in LDS
// (conflict-free column reads), v in LDS.  Backpointers stored as u8.
// Tie-breaking = first-occurrence argmax (strict >, ascending i / q).
// ---------------------------------------------------------------------------
__global__ __launch_bounds__(512) void crf_viterbi_fwd(
    const float* __restrict__ logits, const float* __restrict__ trans,
    unsigned char* __restrict__ bp, float* __restrict__ vlast)
{
    __shared__ float sT[128 * 128];
    __shared__ float sV[128];
    __shared__ float sPV[512];
    __shared__ int   sPI[512];

    const int tid = threadIdx.x;
    const int b   = blockIdx.x;
    const int j   = tid & 127;
    const int q   = tid >> 7;

    for (int i = tid; i < 128 * 128; i += 512) sT[i] = trans[i];
    if (tid < 128) sV[tid] = logits[(b * 512) * 128 + tid];
    __syncthreads();

    for (int t = 1; t < 512; ++t) {
        float best = -3.402823466e38f; int arg = 0;
        const int i0 = q * 32;
#pragma unroll 4
        for (int ii = 0; ii < 32; ++ii) {
            int i = i0 + ii;
            float s = sV[i] + sT[i * 128 + j];   // v[i] broadcast + conflict-free column
            if (s > best) { best = s; arg = i; }
        }
        sPV[tid] = best; sPI[tid] = arg;
        __syncthreads();
        if (tid < 128) {
            float bb = sPV[j]; int aa = sPI[j];
#pragma unroll
            for (int p = 1; p < 4; ++p) {
                float pv = sPV[p * 128 + j];
                if (pv > bb) { bb = pv; aa = sPI[p * 128 + j]; }
            }
            bp[(b * 511 + (t - 1)) * 128 + j] = (unsigned char)aa;
            sV[j] = bb + logits[(b * 512 + t) * 128 + j];
        }
        __syncthreads();
    }
    if (tid < 128) vlast[b * 128 + tid] = sV[tid];
}

// ---------------------------------------------------------------------------
// Kernel 3: backtrace — one thread per batch; bp is L2-resident (4 MB << 192MB)
// ---------------------------------------------------------------------------
__global__ void crf_backtrace(const float* __restrict__ vlast,
                              const unsigned char* __restrict__ bp,
                              float* __restrict__ out)
{
    int b = blockIdx.x * blockDim.x + threadIdx.x;
    if (b >= 64) return;
    float best = -3.402823466e38f; int tag = 0;
    for (int i = 0; i < 128; ++i) {
        float s = vlast[b * 128 + i];
        if (s > best) { best = s; tag = i; }
    }
    out[b * 512 + 511] = (float)tag;
    for (int t = 510; t >= 0; --t) {
        tag = bp[(b * 511 + t) * 128 + tag];
        out[b * 512 + t] = (float)tag;
    }
}

extern "C" void kernel_launch(void* const* d_in, const int* in_sizes, int n_in,
                              void* d_out, int out_size, void* d_ws, size_t ws_size,
                              hipStream_t stream) {
    (void)in_sizes; (void)n_in; (void)out_size; (void)ws_size;
    const float* x     = (const float*)d_in[0];
    const float* w     = (const float*)d_in[1];
    const float* bias  = (const float*)d_in[2];
    const float* trans = (const float*)d_in[3];
    const float* lb    = (const float*)d_in[4];
    const float* rb    = (const float*)d_in[5];
    float* out = (float*)d_out;

    // Workspace: logits 16 MiB | bp 4 MiB (u8, 64*511*128 used) | vlast 32 KiB
    char* ws = (char*)d_ws;
    float*         logits = (float*)ws;
    unsigned char* bp     = (unsigned char*)(ws + (16u << 20));
    float*         vlast  = (float*)(ws + (16u << 20) + (4u << 20));

    crf_gemm_logits<<<2048, 256, 0, stream>>>(x, w, bias, lb, rb, logits);
    crf_viterbi_fwd<<<64, 512, 0, stream>>>(logits, trans, bp, vlast);
    crf_backtrace<<<2, 32, 0, stream>>>(vlast, bp, out);
}
